// DozatLstmCell_34668976013861
// MI455X (gfx1250) — matrix operand order repaired and checked by reading, not compiled, not run on hardware
//
#include <hip/hip_runtime.h>

typedef __bf16 bf16_t;
typedef __attribute__((ext_vector_type(16))) __bf16 v16bf;
typedef __attribute__((ext_vector_type(8)))  __bf16 v8bf;
typedef __attribute__((ext_vector_type(8)))  float   v8f;
typedef __attribute__((ext_vector_type(4)))  unsigned int v4u;
typedef __attribute__((ext_vector_type(8)))  int          v8i;
typedef __attribute__((ext_vector_type(4)))  int          v4i;

#define B_   64
#define T_   512
#define D_   512
#define H_   512
#define KTOT 1024          // D + H (A = [x_t | h])
#define NP   2048          // 4*H output columns, permuted n' = 4*h + gate
#define AROW 520           // bf16 LDS row stride: 512 data + 8 pad (260 dwords -> conflict-free b128)
#define PROW 260           // padded float row stride for proj tile
#define LDSX_OFF 0
#define LDSH_OFF (16 * AROW * 2)            // 16640 B
#define LDSP_OFF (2 * 16 * AROW * 2)        // 33280 B
#define SMEM_BYTES (LDSP_OFF + 16 * PROW * 4)  // 49920 B

// ---------------- preprocessing ----------------

__global__ __launch_bounds__(256) void lstm_prep_w_34668976013861(
    const float* __restrict__ Wi, const float* __restrict__ Wh, bf16_t* __restrict__ Wcat)
{
    int idx = blockIdx.x * 256 + threadIdx.x;           // over NP*KTOT
    if (idx >= NP * KTOT) return;
    int n = idx >> 10, k = idx & 1023;                  // n' = 4*h + g
    int h = n >> 2, g = n & 3;
    float w = (k < D_) ? Wi[(size_t)(g * H_ + h) * D_ + k]
                       : Wh[(size_t)(g * H_ + h) * H_ + (k - D_)];
    Wcat[idx] = (bf16_t)w;
}

__global__ __launch_bounds__(256) void lstm_prep_x_34668976013861(
    const float* __restrict__ x, bf16_t* __restrict__ xbf)
{
    size_t idx = (size_t)blockIdx.x * 256 + threadIdx.x;  // over B*T*D
    if (idx < (size_t)B_ * T_ * D_) xbf[idx] = (bf16_t)x[idx];
}

__global__ __launch_bounds__(256) void lstm_prep_state_34668976013861(
    const float* __restrict__ bi, const float* __restrict__ bh,
    float* __restrict__ bcat, bf16_t* __restrict__ h0, float* __restrict__ c0)
{
    int idx = blockIdx.x * 256 + threadIdx.x;
    if (idx < NP) {
        int h = idx >> 2, g = idx & 3;
        bcat[idx] = bi[g * H_ + h] + bh[g * H_ + h];
    }
    if (idx < B_ * H_) { h0[idx] = (bf16_t)0.f; c0[idx] = 0.f; }
}

// ---------------- TDM helper: 2-D bf16 tile (tile_d0 x tile_d1) global -> LDS ----------------
// Pad config: pad_enable, interval = 256 dwords (one 512-bf16 row), amount = 4 dwords
// => LDS row stride 520 bf16 (260-dword stride -> conflict-free ds_load_b128 fragments).

__device__ __forceinline__ void tdm_load_2d_bf16(
    unsigned int lds_byte_addr, const bf16_t* gptr,
    unsigned int tensor_d0, unsigned int tensor_d1,
    unsigned int tile_d0, unsigned int tile_d1,
    unsigned long long stride0_elems)
{
    unsigned long long ga = (unsigned long long)(uintptr_t)gptr;
    v4u g0;
    g0[0] = 1u;                                              // count=1 (user descriptor)
    g0[1] = lds_byte_addr;                                   // lds_addr
    g0[2] = (unsigned int)ga;                                // global_addr[31:0]
    g0[3] = (unsigned int)((ga >> 32) & 0x01FFFFFFu) | (2u << 30);  // addr[56:32] | type=2
    v8i g1;
    g1[0] = (int)((1u << 16)      // data_size = 1 -> 2 bytes
                | (1u << 20)      // pad_enable
                | (7u << 22)      // pad_interval = 256 dwords
                | (3u << 25));    // pad_amount  = 4 dwords
    g1[1] = (int)((tensor_d0 & 0xFFFFu) << 16);                              // dim0[15:0]
    g1[2] = (int)(((tensor_d0 >> 16) & 0xFFFFu) | ((tensor_d1 & 0xFFFFu) << 16));
    g1[3] = (int)(((tensor_d1 >> 16) & 0xFFFFu) | ((tile_d0 & 0xFFFFu) << 16));
    g1[4] = (int)(tile_d1 & 0xFFFFu);                                        // tile_dim1, tile_dim2=0
    g1[5] = (int)(unsigned int)(stride0_elems & 0xFFFFFFFFu);                // dim0_stride[31:0]
    g1[6] = (int)(unsigned int)((stride0_elems >> 32) & 0xFFFFu);            // dim0_stride[47:32]
    g1[7] = 0;
    v4i z4 = {0, 0, 0, 0};
    v8i z8 = {0, 0, 0, 0, 0, 0, 0, 0};
    __builtin_amdgcn_tensor_load_to_lds(g0, g1, z4, z4, z8, 0);
}

// ---------------- per-timestep fused kernel ----------------

__device__ __forceinline__ float sigf(float x)     { return 1.f / (1.f + __expf(-x)); }
__device__ __forceinline__ float tanhfast(float x) { float e = __expf(2.f * x); return (e - 1.f) / (e + 1.f); }

__global__ __launch_bounds__(256) void lstm_step_34668976013861(
    const bf16_t* __restrict__ xbf, const int* __restrict__ lengths,
    const float* __restrict__ dmask, const bf16_t* __restrict__ Wcat,
    const float* __restrict__ bcat, const bf16_t* __restrict__ hIn,
    bf16_t* __restrict__ hOut, float* __restrict__ cbuf,
    float* __restrict__ out, int t)
{
    __shared__ __align__(16) unsigned char smem[SMEM_BYTES];
    bf16_t* ldsX = (bf16_t*)(smem + LDSX_OFF);   // 16 x 512 (stride 520) bf16
    bf16_t* ldsH = (bf16_t*)(smem + LDSH_OFF);   // 16 x 512 (stride 520) bf16
    float*  ldsP = (float*)(smem + LDSP_OFF);    // 16 x 256 (stride 260) f32 proj tile

    const int tid    = threadIdx.x;
    const int bBase  = blockIdx.x * 16;   // 4 m-tiles over batch
    const int npBase = blockIdx.y * 256;  // 8 n-groups over permuted gate columns
    const int hBase  = blockIdx.y * 64;   // 64 hidden units per workgroup

    // ---- async tile DMA: A = [x_t | h_in] rows via Tensor Data Mover (wave 0 issues)
    if (tid < 32) {
        tdm_load_2d_bf16(LDSX_OFF,
                         xbf + (size_t)bBase * T_ * D_ + (size_t)t * D_,
                         D_, 16, D_, 16, (unsigned long long)T_ * D_);
        tdm_load_2d_bf16(LDSH_OFF,
                         hIn + (size_t)bBase * H_,
                         H_, 16, H_, 16, H_);
        __builtin_amdgcn_s_wait_tensorcnt(0);
    }
    __syncthreads();

    const int wave  = tid >> 5;
    const int lane  = tid & 31;
    const int lrow  = lane & 15;
    const int khalf = lane >> 4;                 // fragment K half select

    // two 16x16 tiles per wave; bias folded into accumulator init (bias depends on n' only)
    const int n0 = npBase + wave * 32 + lrow;
    const int n1 = n0 + 16;
    v8f acc0, acc1;
    {
        float b0 = bcat[n0], b1 = bcat[n1];
        #pragma unroll
        for (int i = 0; i < 8; ++i) { acc0[i] = b0; acc1[i] = b1; }
    }

    const bf16_t* __restrict__ wp0 = Wcat + (size_t)n0 * KTOT;  // row-major [n'][K]
    const bf16_t* __restrict__ wp1 = Wcat + (size_t)n1 * KTOT;
    const bf16_t* apx = ldsX + lrow * AROW + khalf * 16;
    const bf16_t* aph = ldsH + lrow * AROW + khalf * 16;

    // K = 0..511 : x part
    for (int kc = 0; kc < D_; kc += 32) {
        v8bf alo = *(const v8bf*)(apx + kc);
        v8bf ahi = *(const v8bf*)(apx + kc + 8);
        v16bf a;
        #pragma unroll
        for (int i = 0; i < 8; ++i) { a[i] = alo[i]; a[8 + i] = ahi[i]; }
        v16bf b0 = *(const v16bf*)(wp0 + kc + khalf * 16);
        v16bf b1 = *(const v16bf*)(wp1 + kc + khalf * 16);
        acc0 = __builtin_amdgcn_wmma_f32_16x16x32_bf16(false, a, false, b0, (short)0, acc0, false, false);
        acc1 = __builtin_amdgcn_wmma_f32_16x16x32_bf16(false, a, false, b1, (short)0, acc1, false, false);
    }
    // K = 512..1023 : h part
    for (int kc = 0; kc < H_; kc += 32) {
        v8bf alo = *(const v8bf*)(aph + kc);
        v8bf ahi = *(const v8bf*)(aph + kc + 8);
        v16bf a;
        #pragma unroll
        for (int i = 0; i < 8; ++i) { a[i] = alo[i]; a[8 + i] = ahi[i]; }
        v16bf b0 = *(const v16bf*)(wp0 + D_ + kc + khalf * 16);
        v16bf b1 = *(const v16bf*)(wp1 + D_ + kc + khalf * 16);
        acc0 = __builtin_amdgcn_wmma_f32_16x16x32_bf16(false, a, false, b0, (short)0, acc0, false, false);
        acc1 = __builtin_amdgcn_wmma_f32_16x16x32_bf16(false, a, false, b1, (short)0, acc1, false, false);
    }

    // ---- dump proj tiles to LDS (C layout: VGPR v -> row v + 8*khalf, lane&15 -> col)
    {
        int nl = wave * 32 + lrow;
        #pragma unroll
        for (int v = 0; v < 8; ++v) {
            int m = v + (khalf << 3);
            ldsP[m * PROW + nl]      = acc0[v];
            ldsP[m * PROW + nl + 16] = acc1[v];
        }
    }
    __syncthreads();

    // ---- epilogue: gates + cell update for our 16 batches x 64 hidden units
    for (int p = tid; p < 16 * 64; p += 256) {
        int bl = p >> 6, hl = p & 63;
        const float* g4 = &ldsP[bl * PROW + hl * 4];   // n' = 4h+{i,f,g,o} adjacent
        float ig = sigf(g4[0]);
        float fg = sigf(g4[1]);
        float gg = tanhfast(g4[2]);
        float og = sigf(g4[3]);
        int b = bBase + bl, h = hBase + hl;
        float cprev = cbuf[b * H_ + h];
        float cnew  = fmaf(ig, gg, fg * cprev);
        float lr    = (cnew > 0.f) ? cnew : 0.01f * cnew;       // leaky_relu slope 0.01
        float ho    = og * lr * dmask[b * H_ + h];
        bool  act   = t < lengths[b];
        out[(size_t)b * T_ * H_ + (size_t)t * H_ + h] = act ? ho : 0.f;
        cbuf[b * H_ + h] = act ? cnew : cprev;
        hOut[b * H_ + h] = act ? (bf16_t)ho : hIn[b * H_ + h];
    }
}

// ---------------- final h/c: gather h_final from outputs (lengths >= 1), c from cbuf ----------

__global__ __launch_bounds__(256) void lstm_final_34668976013861(
    const float* __restrict__ out, const int* __restrict__ lengths,
    const float* __restrict__ cf, float* __restrict__ dst)
{
    int idx = blockIdx.x * 256 + threadIdx.x;
    if (idx < B_ * H_) {
        int b = idx >> 9, h = idx & (H_ - 1);
        int tl = lengths[b] - 1;
        dst[idx]           = out[(size_t)b * T_ * H_ + (size_t)tl * H_ + h];
        dst[B_ * H_ + idx] = cf[idx];
    }
}

extern "C" void kernel_launch(void* const* d_in, const int* in_sizes, int n_in,
                              void* d_out, int out_size, void* d_ws, size_t ws_size,
                              hipStream_t stream) {
    const float* x       = (const float*)d_in[0];
    const int*   lengths = (const int*)d_in[1];
    const float* dmask   = (const float*)d_in[2];
    const float* Wi      = (const float*)d_in[3];
    const float* bi      = (const float*)d_in[4];
    const float* Wh      = (const float*)d_in[5];
    const float* bh      = (const float*)d_in[6];
    float* out = (float*)d_out;

    // workspace (~36.5 MB): Wcat bf16 | xbf bf16 | bcat f32 | hbf x2 (bf16) | cbuf f32
    uint8_t* ws = (uint8_t*)d_ws;
    bf16_t* Wcat = (bf16_t*)ws;                                     // 4 MiB
    bf16_t* xbf  = (bf16_t*)(ws + (size_t)NP * KTOT * 2);           // 32 MiB
    float*  bcat = (float*)(ws + (size_t)NP * KTOT * 2 + (size_t)B_ * T_ * D_ * 2);
    bf16_t* hbf0 = (bf16_t*)(bcat + NP);
    bf16_t* hbf1 = hbf0 + B_ * H_;
    float*  cbuf = (float*)(hbf1 + B_ * H_);

    lstm_prep_w_34668976013861<<<(NP * KTOT + 255) / 256, 256, 0, stream>>>(Wi, Wh, Wcat);
    lstm_prep_x_34668976013861<<<(B_ * T_ * D_ + 255) / 256, 256, 0, stream>>>(x, xbf);
    lstm_prep_state_34668976013861<<<(B_ * H_ + 255) / 256, 256, 0, stream>>>(bi, bh, bcat, hbf0, cbuf);

    bf16_t* hb[2] = { hbf0, hbf1 };
    dim3 grid(4, 8);   // 4 batch tiles x 8 gate-column groups = 32 WGs/step
    for (int t = 0; t < T_; ++t) {
        lstm_step_34668976013861<<<grid, 256, 0, stream>>>(
            xbf, lengths, dmask, Wcat, bcat, hb[t & 1], hb[(t + 1) & 1], cbuf, out, t);
    }
    lstm_final_34668976013861<<<(B_ * H_ + 255) / 256, 256, 0, stream>>>(
        out, lengths, cbuf, out + (size_t)B_ * T_ * H_);
}
